// SelfAttention2d_20744692040331
// MI455X (gfx1250) — compile-verified, matrix operand order checked by reading, not run on hardware
//
#include <hip/hip_runtime.h>

typedef __bf16 bf16_t;
typedef __attribute__((ext_vector_type(16))) __bf16 v16bf;
typedef __attribute__((ext_vector_type(8)))  __bf16 v8bf;
typedef __attribute__((ext_vector_type(8)))  float  v8f;

union BF16x16 { v16bf v; v8bf h[2]; __bf16 e[16]; };
union F32x8   { v8f v; float e[8]; };

__device__ __forceinline__ v8f wmma_bf16(v16bf a, v16bf b, v8f c) {
  // D = A(16x32 bf16) * B(32x16 bf16) + C(16x16 f32)
  return __builtin_amdgcn_wmma_f32_16x16x32_bf16(
      /*neg_a=*/false, a, /*neg_b=*/false, b,
      /*c_mod=*/(short)0, c, /*reuse_a=*/false, /*reuse_b=*/false);
}

__device__ __forceinline__ bf16_t f2bf(float f) { return (bf16_t)f; }

// gfx1250 async global->LDS copies (ASYNCcnt). Operands per CDNA5 ISA 15.18.3:
// VDST = per-lane LDS byte address, VADDR = 64-bit global address, off = no saddr.
__device__ __forceinline__ void async_load_b128(const void* gaddr, unsigned lds_off) {
  asm volatile("global_load_async_to_lds_b128 %0, %1, off"
               :: "v"(lds_off), "v"(gaddr) : "memory");
}
__device__ __forceinline__ void async_load_b32(const void* gaddr, unsigned lds_off) {
  asm volatile("global_load_async_to_lds_b32 %0, %1, off"
               :: "v"(lds_off), "v"(gaddr) : "memory");
}
__device__ __forceinline__ void wait_async0() {
  asm volatile("s_wait_asynccnt 0x0" ::: "memory");
}

// ---------------------------------------------------------------------------
// Kernel 0: one-shot fp32 -> bf16 weight conversion (Wk,Wq,Wv,Wu -> ws).
// ---------------------------------------------------------------------------
__global__ __launch_bounds__(256) void convert_weights_kernel(
    const float* __restrict__ Wk, const float* __restrict__ Wq,
    const float* __restrict__ Wv, const float* __restrict__ Wu,
    bf16_t* __restrict__ wbf) {
  int idx = blockIdx.x * 256 + threadIdx.x;   // 0..65535
  wbf[idx]          = f2bf(Wk[idx]);
  wbf[65536 + idx]  = f2bf(Wq[idx]);
  wbf[131072 + idx] = f2bf(Wv[idx]);
  wbf[196608 + idx] = f2bf(Wu[idx]);
}

// ---------------------------------------------------------------------------
// Kernel 1: fused QKV projection.  qkv[p][b][o][n] = W_p @ x[b] + b_p  (bf16)
// grid = (16 col-tiles of 64, 32 batches), block = 256 (8 waves)
// ---------------------------------------------------------------------------
__global__ __launch_bounds__(256) void qkv_proj_kernel(
    const float* __restrict__ x,
    const bf16_t* __restrict__ wbf,
    const float* __restrict__ bk, const float* __restrict__ bq,
    const float* __restrict__ bv,
    bf16_t* __restrict__ qkv) {
  extern __shared__ char smem_raw[];
  bf16_t* xs = (bf16_t*)smem_raw;          // [64 cols][256 cin], transposed tile
  const int b   = blockIdx.y;
  const int n0g = blockIdx.x * 64;
  const int tid = threadIdx.x;

  // stage x tile (fp32 -> bf16), transposed so B fragments are contiguous
  const int j  = tid & 63;
  const int cb = tid >> 6;
#pragma unroll 4
  for (int cc = 0; cc < 64; ++cc) {
    int c = cc * 4 + cb;
    float val = x[(((size_t)b * 256 + c) << 10) + n0g + j];
    xs[j * 256 + c] = f2bf(val);
  }
  __syncthreads();

  const int wv    = tid >> 5;
  const int lane  = tid & 31;
  const int mrow  = lane & 15;
  const int lhalf = lane >> 4;
  const int khalf = lhalf * 8;

  for (int q = 0; q < 24; ++q) {
    int tile = q * 8 + wv;        // 0..191
    int cg = tile & 3;            // col sub-tile (16 cols)
    int rt = tile >> 2;           // 0..47
    int p  = rt >> 4;             // projection 0..2
    int r  = rt & 15;             // row tile
    int o0 = r * 16;
    const bf16_t* W = wbf + (size_t)p * 65536;
    const float* bias = (p == 0) ? bk : (p == 1) ? bq : bv;

    F32x8 acc;
#pragma unroll
    for (int i = 0; i < 8; ++i) acc.e[i] = 0.f;

    const int orow = o0 + mrow;
#pragma unroll
    for (int kk = 0; kk < 8; ++kk) {
      int k0 = kk * 32;
      // A fragment (bf16 weights, pre-converted): two 16B loads
      BF16x16 a;
      const bf16_t* wp = W + orow * 256 + k0 + khalf;
      a.h[0] = *(const v8bf*)(wp);
      a.h[1] = *(const v8bf*)(wp + 16);
      // B fragment from LDS (already transposed): contiguous ds_load_b128
      BF16x16 bf;
      const bf16_t* xp = xs + (cg * 16 + mrow) * 256 + k0 + khalf;
      bf.h[0] = *(const v8bf*)(xp);
      bf.h[1] = *(const v8bf*)(xp + 16);
      acc.v = wmma_bf16(a.v, bf.v, acc.v);
    }

    int ncol = n0g + cg * 16 + mrow;
#pragma unroll
    for (int i = 0; i < 8; ++i) {
      int o = o0 + i + 8 * lhalf;   // C-layout row
      float val = acc.e[i] + bias[o];
      qkv[(((size_t)p * 32 + b) * 256 + o) * 1024 + ncol] = f2bf(val);
    }
  }
}

// ---------------------------------------------------------------------------
// Kernel 2: flash attention per (batch, head). K,Q,V fully LDS-resident.
// grid = 256 (b*8+head), block = 256 (8 waves); each wave owns 128 queries.
// V and pos_enc staged with gfx1250 async global->LDS DMA.
// ---------------------------------------------------------------------------
__global__ __launch_bounds__(256) void attention_kernel(
    const bf16_t* __restrict__ qkv,
    const float* __restrict__ pos_enc,
    bf16_t* __restrict__ attn) {
  extern __shared__ char smem_raw[];
  bf16_t* Kt  = (bf16_t*)(smem_raw);            // [1024 keys][32 c]  (K^T)
  bf16_t* Qt  = (bf16_t*)(smem_raw + 65536);    // [1024 qrys][32 c]  (Q^T)
  bf16_t* Vs  = (bf16_t*)(smem_raw + 131072);   // [32 c][1024 keys]
  float*  pos = (float*)(smem_raw + 196608);    // [3969] per-head bias

  const int head = blockIdx.x & 7;
  const int b    = blockIdx.x >> 3;
  const int tid  = threadIdx.x;
  const size_t hb = (((size_t)b) * 256 + head * 32) * 1024;
  const bf16_t* Kg = qkv + hb;
  const bf16_t* Qg = qkv + (size_t)32 * 256 * 1024 + hb;
  const bf16_t* Vg = qkv + (size_t)2 * 32 * 256 * 1024 + hb;

  // async DMA: V (straight 64KB copy) and per-head pos_enc slice
  const unsigned vs_off  = (unsigned)(uintptr_t)(void*)Vs;
  const unsigned pos_off = (unsigned)(uintptr_t)(void*)pos;
  for (int v = tid; v < 4096; v += 256)
    async_load_b128(Vg + v * 8, vs_off + v * 16);
  for (int i = tid; i < 3969; i += 256)
    async_load_b32(pos_enc + head * 3969 + i, pos_off + i * 4);

  // K^T, Q^T staging (transpose: VGPR path)
  for (int v = tid; v < 4096; v += 256) {
    int c  = v >> 7;
    int n0 = (v & 127) * 8;
    v8bf kv = *(const v8bf*)(Kg + c * 1024 + n0);
    v8bf qv = *(const v8bf*)(Qg + c * 1024 + n0);
#pragma unroll
    for (int t = 0; t < 8; ++t) {
      Kt[(n0 + t) * 32 + c] = kv[t];
      Qt[(n0 + t) * 32 + c] = qv[t];
    }
  }
  wait_async0();
  __syncthreads();

  const int wv    = tid >> 5;
  const int lane  = tid & 31;
  const int mloc  = lane & 15;
  const int lhalf = lane >> 4;
  const int khalf = lhalf * 8;
  const float scale = 0.17677669529663687f;   // 32^-0.5

  for (int q = 0; q < 8; ++q) {
    const int m0 = (wv * 8 + q) * 16;
    const int m  = m0 + mloc;
    const int y2 = m >> 5, x2 = m & 31;

    // Q fragment for this m-tile (reused across all key chunks)
    BF16x16 bq;
    { const bf16_t* qp = Qt + m * 32 + khalf;
      bq.h[0] = *(const v8bf*)(qp);
      bq.h[1] = *(const v8bf*)(qp + 16); }

    F32x8 o0a, o1a;
#pragma unroll
    for (int i = 0; i < 8; ++i) { o0a.e[i] = 0.f; o1a.e[i] = 0.f; }
    float mu = -3.0e38f, ell = 0.f;

    for (int nk = 0; nk < 32; ++nk) {       // 32 keys per chunk
      const int n0 = nk * 32;
      // K^T fragments: rows = keys n0..n0+15 and n0+16..n0+31
      BF16x16 a0, a1;
      { const bf16_t* kp = Kt + (n0 + mloc) * 32 + khalf;
        a0.h[0] = *(const v8bf*)(kp);
        a0.h[1] = *(const v8bf*)(kp + 16);
        kp += 16 * 32;
        a1.h[0] = *(const v8bf*)(kp);
        a1.h[1] = *(const v8bf*)(kp + 16); }

      v8f zero;
#pragma unroll
      for (int i = 0; i < 8; ++i) zero[i] = 0.f;
      F32x8 s0, s1;
      s0.v = wmma_bf16(a0.v, bq.v, zero);   // S[n0..+15 , m-tile]
      s1.v = wmma_bf16(a1.v, bq.v, zero);   // S[n0+16..+31 , m-tile]

      // scale + relative position bias, tile column max
      float tmax = -3.0e38f;
#pragma unroll
      for (int i = 0; i < 8; ++i) {
        int n  = n0 + i + 8 * lhalf;
        int pi = ((n >> 5) - y2 + 31) * 63 + ((n & 31) - x2 + 31);
        s0.e[i] = s0.e[i] * scale + pos[pi];
        int n2  = n + 16;
        int pi2 = ((n2 >> 5) - y2 + 31) * 63 + ((n2 & 31) - x2 + 31);
        s1.e[i] = s1.e[i] * scale + pos[pi2];
        tmax = fmaxf(tmax, fmaxf(s0.e[i], s1.e[i]));
      }
      tmax = fmaxf(tmax, __shfl_xor(tmax, 16, 32));  // lanes L / L+16 share col
      float mun   = fmaxf(mu, tmax);
      float alpha = __expf(mu - mun);

      float rs = 0.f;
      BF16x16 bp;                // P repack: no cross-lane movement needed
#pragma unroll
      for (int i = 0; i < 8; ++i) {
        float p0 = __expf(s0.e[i] - mun);
        float p1 = __expf(s1.e[i] - mun);
        rs += p0 + p1;
        bp.e[i]     = f2bf(p0);  // K 0..7 (or 8..15 in upper lanes)
        bp.e[8 + i] = f2bf(p1);  // K 16..23 (or 24..31)
        o0a.e[i] *= alpha;
        o1a.e[i] *= alpha;
      }
      rs += __shfl_xor(rs, 16, 32);
      ell = ell * alpha + rs;
      mu  = mun;

      // V fragments (natural layout, contiguous along keys)
      BF16x16 av0, av1;
      { const bf16_t* vp = Vs + mloc * 1024 + n0 + khalf;
        av0.h[0] = *(const v8bf*)(vp);
        av0.h[1] = *(const v8bf*)(vp + 16);
        vp += 16 * 1024;
        av1.h[0] = *(const v8bf*)(vp);
        av1.h[1] = *(const v8bf*)(vp + 16); }
      o0a.v = wmma_bf16(av0.v, bp.v, o0a.v);  // O rows c=0..15
      o1a.v = wmma_bf16(av1.v, bp.v, o1a.v);  // O rows c=16..31
    }

    float inv = 1.f / ell;
#pragma unroll
    for (int i = 0; i < 8; ++i) {
      int cr = i + 8 * lhalf;
      attn[((size_t)b * 256 + head * 32 + cr) * 1024 + m]      = f2bf(o0a.e[i] * inv);
      attn[((size_t)b * 256 + head * 32 + 16 + cr) * 1024 + m] = f2bf(o1a.e[i] * inv);
    }
  }
}

// ---------------------------------------------------------------------------
// Kernel 3: output projection.  out = Wu @ attn + bu  (fp32 out)
// ---------------------------------------------------------------------------
__global__ __launch_bounds__(256) void out_proj_kernel(
    const bf16_t* __restrict__ attn,
    const bf16_t* __restrict__ Wu, const float* __restrict__ bu,
    float* __restrict__ out) {
  extern __shared__ char smem_raw[];
  bf16_t* xs = (bf16_t*)smem_raw;          // [64 cols][256 c], transposed tile
  const int b   = blockIdx.y;
  const int n0g = blockIdx.x * 64;
  const int tid = threadIdx.x;

  for (int v = tid; v < 2048; v += 256) {
    int c  = v >> 3;
    int j0 = (v & 7) * 8;
    v8bf av = *(const v8bf*)(attn + ((size_t)b * 256 + c) * 1024 + n0g + j0);
#pragma unroll
    for (int t = 0; t < 8; ++t) xs[(j0 + t) * 256 + c] = av[t];
  }
  __syncthreads();

  const int wv    = tid >> 5;
  const int lane  = tid & 31;
  const int mrow  = lane & 15;
  const int lhalf = lane >> 4;
  const int khalf = lhalf * 8;

  for (int q = 0; q < 8; ++q) {
    int tile = wv * 8 + q;       // 0..63
    int cg = tile & 3;
    int r  = tile >> 2;
    int o0 = r * 16;
    F32x8 acc;
#pragma unroll
    for (int i = 0; i < 8; ++i) acc.e[i] = 0.f;

    const int orow = o0 + mrow;
#pragma unroll
    for (int kk = 0; kk < 8; ++kk) {
      int k0 = kk * 32;
      BF16x16 a;
      const bf16_t* wp = Wu + orow * 256 + k0 + khalf;
      a.h[0] = *(const v8bf*)(wp);
      a.h[1] = *(const v8bf*)(wp + 16);
      BF16x16 bf;
      const bf16_t* xp = xs + (cg * 16 + mrow) * 256 + k0 + khalf;
      bf.h[0] = *(const v8bf*)(xp);
      bf.h[1] = *(const v8bf*)(xp + 16);
      acc.v = wmma_bf16(a.v, bf.v, acc.v);
    }

    int ncol = n0g + cg * 16 + mrow;
#pragma unroll
    for (int i = 0; i < 8; ++i) {
      int o = o0 + i + 8 * lhalf;
      out[((size_t)b * 256 + o) * 1024 + ncol] = acc.e[i] + bu[o];
    }
  }
}

// ---------------------------------------------------------------------------
extern "C" void kernel_launch(void* const* d_in, const int* in_sizes, int n_in,
                              void* d_out, int out_size, void* d_ws, size_t ws_size,
                              hipStream_t stream) {
  const float* x   = (const float*)d_in[0];
  const float* Wk  = (const float*)d_in[1];
  const float* bk  = (const float*)d_in[2];
  const float* Wq  = (const float*)d_in[3];
  const float* bq  = (const float*)d_in[4];
  const float* Wv  = (const float*)d_in[5];
  const float* bv  = (const float*)d_in[6];
  const float* Wu  = (const float*)d_in[7];
  const float* bu  = (const float*)d_in[8];
  const float* pos = (const float*)d_in[9];
  float* out = (float*)d_out;

  // workspace: qkv (48MB bf16) + attn (16MB bf16) + wbf (512KB bf16)
  bf16_t* qkv  = (bf16_t*)d_ws;
  bf16_t* attn = qkv + (size_t)3 * 32 * 256 * 1024;
  bf16_t* wbf  = attn + (size_t)32 * 256 * 1024;

  dim3 blk(256);
  dim3 g1(16, 32);
  convert_weights_kernel<<<dim3(256), blk, 0, stream>>>(Wk, Wq, Wv, Wu, wbf);
  qkv_proj_kernel<<<g1, blk, 64 * 256 * 2, stream>>>(x, wbf, bk, bq, bv, qkv);
  attention_kernel<<<dim3(256), blk, 212992, stream>>>(qkv, pos, attn);
  out_proj_kernel<<<g1, blk, 64 * 256 * 2, stream>>>(attn, wbf + 196608, bu, out);
}
